// MaskNet_13237089206819
// MI455X (gfx1250) — matrix operand order repaired and checked by reading, not compiled, not run on hardware
//
#include <hip/hip_runtime.h>
#include <hip/hip_bf16.h>

// ---------------------------------------------------------------------------
// MaskNet (Conv-TasNet style) for MI455X / gfx1250, wave32 + WMMA bf16.
// GEMMs: v_wmma_f32_16x16x32_bf16, 16x64 tile per wave (A-fragment reuse x4),
// B strips staged to LDS via GLOBAL_LOAD_ASYNC_TO_LDS_B128 when available.
// gLN statistics fused into GEMM / deform epilogues (wave reduce + atomics).
// ---------------------------------------------------------------------------

typedef __bf16 bf16_t;
typedef __attribute__((ext_vector_type(16))) __bf16 v16bf;
typedef __attribute__((ext_vector_type(8)))  __bf16 v8bf;
typedef __attribute__((ext_vector_type(8)))  float  v8f;
typedef __attribute__((ext_vector_type(4)))  int    v4i;

#define M_SM 4
#define K_T  2000
#define N_CH 512
#define B_D  128
#define H_D  512
#define C_SPK 2
#define ROWS (M_SM * K_T)          // 8000, divisible by 16 and 64
#define GLN_CNT 1024000.0f         // K_T*N_CH == K_T*H_D
#define EPS_F 1e-8f

#ifndef __has_builtin
#define __has_builtin(x) 0
#endif
#if defined(__AMDGCN__) && \
    __has_builtin(__builtin_amdgcn_global_load_async_to_lds_b128) && \
    __has_builtin(__builtin_amdgcn_s_wait_asynccnt)
#define USE_ASYNC_LDS 1
// Builtin signature (from hipcc diagnostic): param0 = v4i in addrspace(1),
// param1 = v4i in addrspace(3), then imm offset + imm cpol.
#define ASYNC_G(p) ((__attribute__((address_space(1))) v4i*)(p))
#define ASYNC_L(p) ((__attribute__((address_space(3))) v4i*)(p))
#else
#define USE_ASYNC_LDS 0
#endif

__device__ __forceinline__ float warp_sum(float v) {
  #pragma unroll
  for (int m = 16; m > 0; m >>= 1) v += __shfl_xor(v, m, 32);
  return v;
}

// ---------------------------------------------------------------------------
// Weight transpose + fp32 -> bf16 convert:  src[rows][cols] -> dst[cols][rows]
// ---------------------------------------------------------------------------
__global__ void wconv_k(const float* __restrict__ src, bf16_t* __restrict__ dst,
                        int rows, int cols) {
  int i = blockIdx.x * blockDim.x + threadIdx.x;
  if (i >= rows * cols) return;
  int r = i / cols, c = i % cols;
  dst[(size_t)c * rows + r] = (bf16_t)src[i];
}

// All 24 blocks' w1/w2 in one launch; pointer table passed by value.
struct BlkW { const float* w1[24]; const float* w2[24]; };

__global__ void wconv_blocks_k(BlkW wp, bf16_t* __restrict__ dst) {
  int i = blockIdx.x * blockDim.x + threadIdx.x;     // over 24*2*65536
  if (i >= 24 * 2 * B_D * H_D) return;
  int e = i & (B_D * H_D - 1);                       // 65536 elements / weight
  int w = i >> 16;                                   // weight index 0..47
  int blkI = w >> 1;
  int second = w & 1;
  const float* src = second ? wp.w2[blkI] : wp.w1[blkI];
  int rows = second ? H_D : B_D;                     // w1:[B][H], w2:[H][B]
  int cols = second ? B_D : H_D;
  int r = e / cols, c = e % cols;
  dst[(size_t)w * (B_D * H_D) + (size_t)c * rows + r] = (bf16_t)src[e];
}

__global__ void zero_k(float* p, int n) {
  int i = blockIdx.x * blockDim.x + threadIdx.x;
  if (i < n) p[i] = 0.f;
}

// ---------------------------------------------------------------------------
// Input gLN statistics: sum / sumsq per sample m over N*K elements.
// ---------------------------------------------------------------------------
__global__ __launch_bounds__(256)
void reduce_in_k(const float* __restrict__ x, float* __restrict__ stats) {
  int m = blockIdx.y;
  const float* xm = x + (size_t)m * N_CH * K_T;
  float s = 0.f, ss = 0.f;
  for (int i = blockIdx.x * blockDim.x + threadIdx.x; i < N_CH * K_T;
       i += gridDim.x * blockDim.x) {
    float v = xm[i]; s += v; ss += v * v;
  }
  s = warp_sum(s); ss = warp_sum(ss);
  __shared__ float sh[16];
  int lane = threadIdx.x & 31, wv = threadIdx.x >> 5;
  if (lane == 0) { sh[wv] = s; sh[wv + 8] = ss; }
  __syncthreads();
  if (threadIdx.x == 0) {
    float ts = 0.f, tss = 0.f;
    #pragma unroll
    for (int i = 0; i < 8; ++i) { ts += sh[i]; tss += sh[i + 8]; }
    atomicAdd(&stats[2 * m], ts);
    atomicAdd(&stats[2 * m + 1], tss);
  }
}

// ---------------------------------------------------------------------------
// Input normalize + transpose: x[M][N][K] -> ynorm_bf16[(m*K+k)*N + n]
// ---------------------------------------------------------------------------
__global__ void norm_in_k(const float* __restrict__ x, const float* __restrict__ stats,
                          const float* __restrict__ g, const float* __restrict__ b,
                          bf16_t* __restrict__ yout) {
  size_t i = (size_t)blockIdx.x * blockDim.x + threadIdx.x;
  if (i >= (size_t)M_SM * N_CH * K_T) return;
  int k = (int)(i % K_T);
  int n = (int)((i / K_T) % N_CH);
  int m = (int)(i / ((size_t)K_T * N_CH));
  float mean = stats[2 * m] * (1.0f / GLN_CNT);
  float var  = stats[2 * m + 1] * (1.0f / GLN_CNT) - mean * mean;
  float rstd = rsqrtf(var + EPS_F);
  float v = (x[i] - mean) * rstd * g[n] + b[n];
  yout[((size_t)m * K_T + k) * N_CH + n] = (bf16_t)v;
}

// ---------------------------------------------------------------------------
// gLN apply: out = (x - mean)*rstd*g[ch] + b[ch]; fp32 and/or bf16 destination.
// ---------------------------------------------------------------------------
__global__ void gln_k(const float* __restrict__ xin, const float* __restrict__ stats,
                      const float* __restrict__ g, const float* __restrict__ b,
                      float* __restrict__ outF, bf16_t* __restrict__ outB) {
  size_t i = (size_t)blockIdx.x * blockDim.x + threadIdx.x;
  if (i >= (size_t)ROWS * H_D) return;
  int h = (int)(i % H_D);
  int row = (int)(i / H_D);
  int m = row / K_T;
  float mean = stats[2 * m] * (1.0f / GLN_CNT);
  float var  = stats[2 * m + 1] * (1.0f / GLN_CNT) - mean * mean;
  float rstd = rsqrtf(var + EPS_F);
  float v = (xin[i] - mean) * rstd * g[h] + b[h];
  if (outF) outF[i] = v;
  if (outB) outB[i] = (bf16_t)v;
}

// ---------------------------------------------------------------------------
// Offset head: z[h] = prelu(sum_q hN[m,k+(q-1)d,h]*w_off_d[h,q], a_off)
//              off[m,k,p] = sum_h z[h]*w_off_p[h,p]          (one block per m,k)
// ---------------------------------------------------------------------------
__global__ __launch_bounds__(256)
void offset_k(const float* __restrict__ hN, const float* __restrict__ wod,
              const float* __restrict__ aoff_p, const float* __restrict__ wop,
              float* __restrict__ off, int dil) {
  int bid = blockIdx.x;            // row = m*K + k
  int m = bid / K_T, k = bid % K_T;
  float aoff = *aoff_p;
  float p0 = 0.f, p1 = 0.f, p2 = 0.f;
  for (int h = threadIdx.x; h < H_D; h += 256) {
    float z = 0.f;
    #pragma unroll
    for (int q = 0; q < 3; ++q) {
      int kk = k + (q - 1) * dil;
      if (kk >= 0 && kk < K_T)
        z += hN[((size_t)m * K_T + kk) * H_D + h] * wod[h * 3 + q];
    }
    z = (z >= 0.f) ? z : aoff * z;
    p0 += z * wop[h * 3 + 0];
    p1 += z * wop[h * 3 + 1];
    p2 += z * wop[h * 3 + 2];
  }
  p0 = warp_sum(p0); p1 = warp_sum(p1); p2 = warp_sum(p2);
  __shared__ float sh[24];
  int lane = threadIdx.x & 31, wv = threadIdx.x >> 5;
  if (lane == 0) { sh[wv] = p0; sh[wv + 8] = p1; sh[wv + 16] = p2; }
  __syncthreads();
  if (threadIdx.x == 0) {
    float a = 0.f, b = 0.f, c = 0.f;
    #pragma unroll
    for (int i = 0; i < 8; ++i) { a += sh[i]; b += sh[i + 8]; c += sh[i + 16]; }
    off[(size_t)bid * 3 + 0] = a;
    off[(size_t)bid * 3 + 1] = b;
    off[(size_t)bid * 3 + 2] = c;
  }
}

// ---------------------------------------------------------------------------
// Deformable depthwise conv: bilinear sample along K + tap combine + bias,
// then PReLU(a2) and fused gLN statistics. grid = (ROWS, H/256).
// ---------------------------------------------------------------------------
__global__ __launch_bounds__(256)
void deform_k(const float* __restrict__ hN, const float* __restrict__ off,
              const float* __restrict__ wdw, const float* __restrict__ bdw,
              const float* __restrict__ a2_p, float* __restrict__ outF,
              float* __restrict__ stats, int dil) {
  int row = blockIdx.x;
  int h = blockIdx.y * blockDim.x + threadIdx.x;
  int m = row / K_T, k = row % K_T;
  float a2 = *a2_p;
  float accv = bdw[h];
  const float* base = hN + (size_t)m * K_T * H_D + h;
  #pragma unroll
  for (int p = 0; p < 3; ++p) {
    float pos = (float)(k + (p - 1) * dil) + off[(size_t)row * 3 + p];
    float f0 = floorf(pos);
    float fr = pos - f0;
    int i0 = (int)f0;
    int i1 = i0 + 1;
    int c0 = i0 < 0 ? 0 : (i0 > K_T - 1 ? K_T - 1 : i0);
    int c1 = i1 < 0 ? 0 : (i1 > K_T - 1 ? K_T - 1 : i1);
    float v0 = (i0 >= 0 && i0 < K_T) ? base[(size_t)c0 * H_D] : 0.f;
    float v1 = (i1 >= 0 && i1 < K_T) ? base[(size_t)c1 * H_D] : 0.f;
    accv += (v0 * (1.f - fr) + v1 * fr) * wdw[h * 3 + p];
  }
  float y = (accv >= 0.f) ? accv : a2 * accv;
  outF[(size_t)row * H_D + h] = y;
  float s = warp_sum(y), ss = warp_sum(y * y);
  __shared__ float sh[16];
  int lane = threadIdx.x & 31, wv = threadIdx.x >> 5;
  if (lane == 0) { sh[wv] = s; sh[wv + 8] = ss; }
  __syncthreads();
  if (threadIdx.x == 0) {
    float ts = 0.f, tss = 0.f;
    #pragma unroll
    for (int i = 0; i < 8; ++i) { ts += sh[i]; tss += sh[i + 8]; }
    atomicAdd(&stats[2 * m], ts);
    atomicAdd(&stats[2 * m + 1], tss);
  }
}

// ---------------------------------------------------------------------------
// WMMA bf16 GEMM: C[ROWS][Ncols] = A[ROWS][Kd] * B[Kd][Ncols], B given
// transposed as BT[Ncols][Kd] bf16.
// Block = 4 waves; block tile = 64 rows x 64 cols; wave tile = 16 rows x 64
// cols (4 accumulators, A fragment reused across 4 WMMAs). The 64-col B strip
// is staged to LDS, via async memory->LDS copies (ASYNCcnt) when available.
// Epilogue modes:
//   0: outF = acc,            outB = bf16(acc)          (bottleneck)
//   1: v = prelu(acc,*alpha); outF = v; fused gLN stats (block GEMM1)
//   2: v = acc + resF;        outF = v; outB = bf16(v)  (block GEMM2 residual)
//   3: relu(acc) scattered to (C, M, N, K) layout       (final mask)
// ---------------------------------------------------------------------------
__global__ __launch_bounds__(128)
void gemm_bf16_k(const bf16_t* __restrict__ A, const bf16_t* __restrict__ BT,
                 int Kd, int mode, const float* __restrict__ alpha_ptr,
                 float* __restrict__ outF, bf16_t* __restrict__ outB,
                 const float* __restrict__ resF, float* __restrict__ stats,
                 float* __restrict__ outScatter, int Ncols) {
  __shared__ bf16_t lsB[64 * 512];           // 64 cols x Kd<=512: 64KB max
  const int lane = threadIdx.x & 31;
  const int wv = threadIdx.x >> 5;
  const int tile_row = blockIdx.x * 4 + wv;

  {  // stage the 64-column B strip into LDS
    const v8bf* src = (const v8bf*)(BT + (size_t)blockIdx.y * 64 * Kd);
    v8bf* dst = (v8bf*)lsB;
    int total = (64 * Kd) / 8;
#if USE_ASYNC_LDS
    for (int i = threadIdx.x; i < total; i += blockDim.x)
      __builtin_amdgcn_global_load_async_to_lds_b128(
          ASYNC_G((void*)(src + i)), ASYNC_L((void*)(dst + i)), 0, 0);
    __builtin_amdgcn_s_wait_asynccnt(0);
#else
    for (int i = threadIdx.x; i < total; i += blockDim.x) dst[i] = src[i];
#endif
  }
  __syncthreads();

  v8f acc[4] = {{}, {}, {}, {}};
  const int r0 = tile_row * 16 + (lane & 15);
  const int ka = (lane >> 4) << 3;           // A sub-K offset: 0 or 8
  const int kb = (lane >> 4) << 3;           // B sub-K offset: 0 or 8
  const bf16_t* Arow = A + (size_t)r0 * Kd;

  for (int k0 = 0; k0 < Kd; k0 += 32) {
    v8bf alo = *(const v8bf*)(Arow + k0 + ka);
    v8bf ahi = *(const v8bf*)(Arow + k0 + ka + 16);
    if (k0 + 32 < Kd) __builtin_prefetch(Arow + k0 + 32 + ka, 0, 3);
    v16bf av = __builtin_shufflevector(alo, ahi, 0, 1, 2, 3, 4, 5, 6, 7,
                                       8, 9, 10, 11, 12, 13, 14, 15);
    #pragma unroll
    for (int c = 0; c < 4; ++c) {           // reuse A across 4 column tiles
      const bf16_t* Bc = lsB + ((size_t)(c * 16 + (lane & 15))) * Kd + k0 + kb;
      v8bf blo = *(const v8bf*)(Bc);
      v8bf bhi = *(const v8bf*)(Bc + 16);
      v16bf bv = __builtin_shufflevector(blo, bhi, 0, 1, 2, 3, 4, 5, 6, 7,
                                         8, 9, 10, 11, 12, 13, 14, 15);
      acc[c] = __builtin_amdgcn_wmma_f32_16x16x32_bf16(
          false, av, false, bv, (short)0, acc[c], false, false);
    }
  }

  const float alpha = (mode == 1) ? *alpha_ptr : 0.f;
  float ssum = 0.f, ssq = 0.f;
  #pragma unroll
  for (int c = 0; c < 4; ++c) {
    int colg = blockIdx.y * 64 + c * 16 + (lane & 15);
    #pragma unroll
    for (int v = 0; v < 8; ++v) {
      int rg = tile_row * 16 + v + ((lane >> 4) << 3);  // C layout: M=v (+8 hi)
      size_t idx = (size_t)rg * Ncols + colg;
      float x = acc[c][v];
      if (mode == 0) {
        outF[idx] = x; outB[idx] = (bf16_t)x;
      } else if (mode == 1) {
        float y = (x >= 0.f) ? x : alpha * x;
        outF[idx] = y; ssum += y; ssq += y * y;
      } else if (mode == 2) {
        float y = x + resF[idx];
        outF[idx] = y; outB[idx] = (bf16_t)y;
      } else {
        float y = (x > 0.f) ? x : 0.f;
        int m = rg / K_T, k = rg % K_T;
        int cc = colg >> 9, n = colg & 511;
        outScatter[(((size_t)cc * M_SM + m) * N_CH + n) * K_T + k] = y;
      }
    }
  }
  if (mode == 1) {
    ssum = warp_sum(ssum); ssq = warp_sum(ssq);
    if (lane == 0) {
      int m = (tile_row * 16) / K_T;     // K_T % 16 == 0 -> tile within one m
      atomicAdd(&stats[2 * m], ssum);
      atomicAdd(&stats[2 * m + 1], ssq);
    }
  }
}

// ---------------------------------------------------------------------------
// Host orchestration. Input flat order (jax pytree, dict keys sorted):
//   [0] mixture_w, then 24 blocks x {a1,a2,a_off,b1,b2,b_dw,g1,g2,w1,w2,
//   w_dw,w_off_d,w_off_p}, then ln_b, ln_g, w_bottle, w_mask.
// ---------------------------------------------------------------------------
extern "C" void kernel_launch(void* const* d_in, const int* in_sizes, int n_in,
                              void* d_out, int out_size, void* d_ws, size_t ws_size,
                              hipStream_t stream) {
  (void)in_sizes; (void)n_in; (void)out_size; (void)ws_size;
  const float* mixture = (const float*)d_in[0];
  auto blk = [&](int i, int j) { return (const float*)d_in[1 + i * 13 + j]; };
  const int pbase = 1 + 24 * 13;
  const float* ln_b = (const float*)d_in[pbase + 0];
  const float* ln_g = (const float*)d_in[pbase + 1];
  const float* w_bottle = (const float*)d_in[pbase + 2];
  const float* w_mask = (const float*)d_in[pbase + 3];

  char* ws = (char*)d_ws;
  size_t o = 0;
  auto carve = [&](size_t bytes) -> char* {
    char* p = ws + o; o = (o + bytes + 255) & ~(size_t)255; return p;
  };
  float*  yf    = (float*)carve((size_t)ROWS * B_D * 4);          // y fp32
  bf16_t* yb    = (bf16_t*)carve((size_t)ROWS * B_D * 2);         // y bf16
  float*  hf    = (float*)carve((size_t)ROWS * H_D * 4);          // hidden fp32
  float*  df    = (float*)carve((size_t)ROWS * H_D * 4);          // deform fp32
  bf16_t* db    = (bf16_t*)carve((size_t)ROWS * H_D * 2);         // bf16 (also ynorm)
  float*  offb  = (float*)carve((size_t)ROWS * 3 * 4);            // offsets
  float*  stats = (float*)carve(256);                             // per-m sum/sumsq
  bf16_t* wbotT  = (bf16_t*)carve((size_t)N_CH * B_D * 2);        // [128][512]
  bf16_t* wmaskT = (bf16_t*)carve((size_t)B_D * C_SPK * N_CH * 2);// [1024][128]
  bf16_t* wblkT  = (bf16_t*)carve((size_t)24 * 2 * B_D * H_D * 2);

  // --- weight transpose + bf16 conversion (cheap, deterministic each call) ---
  wconv_k<<<(N_CH * B_D + 255) / 256, 256, 0, stream>>>(w_bottle, wbotT, N_CH, B_D);
  wconv_k<<<(B_D * C_SPK * N_CH + 255) / 256, 256, 0, stream>>>(w_mask, wmaskT, B_D, C_SPK * N_CH);
  {
    BlkW wp;
    for (int i = 0; i < 24; ++i) { wp.w1[i] = blk(i, 8); wp.w2[i] = blk(i, 9); }
    int tot = 24 * 2 * B_D * H_D;
    wconv_blocks_k<<<(tot + 255) / 256, 256, 0, stream>>>(wp, wblkT);
  }

  // --- input gLN + transpose to bf16 ---
  zero_k<<<1, 32, 0, stream>>>(stats, 8);
  reduce_in_k<<<dim3(512, M_SM), 256, 0, stream>>>(mixture, stats);
  size_t tot_in = (size_t)M_SM * N_CH * K_T;
  norm_in_k<<<(unsigned)((tot_in + 255) / 256), 256, 0, stream>>>(mixture, stats, ln_g, ln_b, db);

  // --- bottleneck GEMM: [8000,512]x[512,128] -> y ---
  gemm_bf16_k<<<dim3(ROWS / 64, B_D / 64), 128, 0, stream>>>(
      db, wbotT, N_CH, 0, nullptr, yf, yb, nullptr, nullptr, nullptr, B_D);

  size_t totH = (size_t)ROWS * H_D;
  for (int i = 0; i < 24; ++i) {
    int dil = 1 << (i & 7);
    bf16_t* w1T = wblkT + (size_t)i * 2 * B_D * H_D;
    bf16_t* w2T = w1T + (size_t)B_D * H_D;

    // h = prelu(y @ w1, a1) + fused gLN stats
    zero_k<<<1, 32, 0, stream>>>(stats, 8);
    gemm_bf16_k<<<dim3(ROWS / 64, H_D / 64), 128, 0, stream>>>(
        yb, w1T, B_D, 1, blk(i, 0), hf, nullptr, nullptr, stats, nullptr, H_D);
    // h = gLN(h, g1, b1) in-place fp32
    gln_k<<<(unsigned)((totH + 255) / 256), 256, 0, stream>>>(
        hf, stats, blk(i, 6), blk(i, 3), hf, nullptr);
    // per-frame deformable offsets
    offset_k<<<ROWS, 256, 0, stream>>>(hf, blk(i, 11), blk(i, 2), blk(i, 12), offb, dil);
    // deform sample + dw combine + prelu(a2) + fused gLN stats
    zero_k<<<1, 32, 0, stream>>>(stats, 8);
    deform_k<<<dim3(ROWS, H_D / 256), 256, 0, stream>>>(
        hf, offb, blk(i, 10), blk(i, 5), blk(i, 1), df, stats, dil);
    // gLN(g2,b2) -> bf16 for next GEMM
    gln_k<<<(unsigned)((totH + 255) / 256), 256, 0, stream>>>(
        df, stats, blk(i, 7), blk(i, 4), nullptr, db);
    // y = res + h2 @ w2  (residual fused in epilogue, writes fp32 + bf16)
    gemm_bf16_k<<<dim3(ROWS / 64, B_D / 64), 128, 0, stream>>>(
        db, w2T, H_D, 2, nullptr, yf, yb, yf, nullptr, nullptr, B_D);
  }

  // --- final mask GEMM + ReLU + (C,M,N,K) scatter ---
  gemm_bf16_k<<<dim3(ROWS / 64, (C_SPK * N_CH) / 64), 128, 0, stream>>>(
      yb, wmaskT, B_D, 3, nullptr, nullptr, nullptr, nullptr, nullptr,
      (float*)d_out, C_SPK * N_CH);
}
